// TimeSeriesAttention_49581102465784
// MI455X (gfx1250) — compile-verified, matrix operand order checked by reading, not run on hardware
//
#include <hip/hip_runtime.h>
#include <stddef.h>
#include <stdint.h>

typedef __bf16 bf16_t;
typedef __attribute__((ext_vector_type(16))) __bf16 v16bf;
typedef __attribute__((ext_vector_type(8)))  __bf16 v8bf;
typedef __attribute__((ext_vector_type(8)))  float  v8f;
typedef __attribute__((ext_vector_type(4)))  float  v4f;

#define S_LEN 4096
#define HDIM  256
#define BATCH 4

// ---------- helpers ----------

__device__ __forceinline__ v16bf cat16(v8bf lo, v8bf hi) {
  return __builtin_shufflevector(lo, hi, 0,1,2,3,4,5,6,7,8,9,10,11,12,13,14,15);
}

__device__ __forceinline__ v8bf load8bf(const bf16_t* p) {
  return *(const v8bf*)p;   // 16B -> one b128 / ds_load_b128
}

__device__ __forceinline__ v8bf cvt8(const float* p) {
  v4f a = *(const v4f*)p;
  v4f b = *(const v4f*)(p + 4);
  v8bf r;
  r[0] = (bf16_t)a[0]; r[1] = (bf16_t)a[1]; r[2] = (bf16_t)a[2]; r[3] = (bf16_t)a[3];
  r[4] = (bf16_t)b[0]; r[5] = (bf16_t)b[1]; r[6] = (bf16_t)b[2]; r[7] = (bf16_t)b[3];
  return r;
}

__device__ __forceinline__ v8f wmma_bf16(v16bf a, v16bf b, v8f c) {
  // D = A(16x32 bf16) * B(32x16 bf16) + C(16x16 f32)
  return __builtin_amdgcn_wmma_f32_16x16x32_bf16(
      false, a, false, b, (short)0, c, false, false);
}

// gfx1250 async global->LDS copy, tracked by ASYNCcnt (CDNA5 ISA).
// vdst = per-lane 32-bit LDS byte offset, vaddr = per-lane 64-bit global addr (GV mode).
__device__ __forceinline__ void async_copy_b128(uint32_t lds_off, const void* gsrc) {
  asm volatile("global_load_async_to_lds_b128 %0, %1, off"
               :: "v"(lds_off), "v"(gsrc) : "memory");
}

// Async loads complete in order per wave, so a partial wait releases the
// oldest stage while the newest stays in flight (pipelining).
template <int N>
__device__ __forceinline__ void wait_async() {
  asm volatile("s_wait_asynccnt %0" :: "i"(N) : "memory");
}

// addrspace(3) offset == low 32 bits of the generic pointer (aperture in high word)
__device__ __forceinline__ uint32_t lds_addr(const void* p) {
  return (uint32_t)(uintptr_t)p;
}

// ---------- kernel 0: pack weights W[in,out] f32 -> Wt[out,in] bf16 ----------

__global__ void pack_weights(const float* __restrict__ Wq,
                             const float* __restrict__ Wk,
                             const float* __restrict__ Wv,
                             bf16_t* __restrict__ Wt) {
  int idx = blockIdx.x * 256 + threadIdx.x;     // 3 * 256 * 256 threads
  int mat = idx >> 16;
  int r   = idx & 65535;
  int o   = r >> 8;      // output feature
  int i   = r & 255;     // input feature
  const float* W = (mat == 0) ? Wq : ((mat == 1) ? Wk : Wv);
  Wt[mat * 65536 + o * 256 + i] = (bf16_t)W[i * 256 + o];
}

// ---------- kernel 1: QKV projection (one wave = one 16x256 output strip) ----------

__global__ void __launch_bounds__(256) qkv_proj(
    const float*  __restrict__ x,   // [B*S, H] f32
    const bf16_t* __restrict__ Wt,  // [3][H_out, H_in] bf16
    const float*  __restrict__ bq,
    const float*  __restrict__ bk,
    const float*  __restrict__ bv,
    bf16_t* __restrict__ Q,         // [B*S, H] bf16 (pre-scaled by 1/16)
    bf16_t* __restrict__ K,         // [B*S, H] bf16
    bf16_t* __restrict__ Vt) {      // [B][H][S] bf16 (transposed V)
  const int lane = threadIdx.x & 31;
  const int wave = threadIdx.x >> 5;
  const int l16  = lane & 15;
  const int half = lane >> 4;       // 0: K-chunk base 0, 1: K-chunk base 8
  const int c0   = half * 8;

  int tile  = blockIdx.x * 8 + wave;     // 0..3071
  int mat   = tile >> 10;                // 0=Q 1=K 2=V
  int mtile = tile & 1023;               // 0..1023
  int row0  = mtile * 16;

  const bf16_t* W  = Wt + mat * 65536;
  const float*  xr = x + (size_t)(row0 + l16) * HDIM;   // A: lane = row M

  v8f zero = {};
  v8f acc[16];
#pragma unroll
  for (int u = 0; u < 16; ++u) acc[u] = zero;

#pragma unroll
  for (int kc = 0; kc < 8; ++kc) {
    int k0 = kc * 32;
    v16bf A = cat16(cvt8(xr + k0 + c0), cvt8(xr + k0 + 16 + c0));
#pragma unroll
    for (int u = 0; u < 16; ++u) {
      const bf16_t* wr = W + (size_t)(u * 16 + l16) * HDIM + k0;  // B: lane = col N
      v16bf Bm = cat16(load8bf(wr + c0), load8bf(wr + 16 + c0));
      acc[u] = wmma_bf16(A, Bm, acc[u]);
    }
  }

  const float* bias = (mat == 0) ? bq : ((mat == 1) ? bk : bv);
  if (mat == 0) {
#pragma unroll
    for (int u = 0; u < 16; ++u) {
      float bcol = bias[u * 16 + l16];
#pragma unroll
      for (int v = 0; v < 8; ++v) {
        int m = v + half * 8;             // C layout: VGPR v -> row m
        Q[(size_t)(row0 + m) * HDIM + u * 16 + l16] =
            (bf16_t)((acc[u][v] + bcol) * 0.0625f);   // fold 1/sqrt(H)
      }
    }
  } else if (mat == 1) {
#pragma unroll
    for (int u = 0; u < 16; ++u) {
      float bcol = bias[u * 16 + l16];
#pragma unroll
      for (int v = 0; v < 8; ++v) {
        int m = v + half * 8;
        K[(size_t)(row0 + m) * HDIM + u * 16 + l16] = (bf16_t)(acc[u][v] + bcol);
      }
    }
  } else {
#pragma unroll
    for (int u = 0; u < 16; ++u) {
      float bcol = bias[u * 16 + l16];
#pragma unroll
      for (int v = 0; v < 8; ++v) {
        int g = row0 + v + half * 8;
        int b = g >> 12;
        int s = g & (S_LEN - 1);
        Vt[(size_t)b * HDIM * S_LEN + (size_t)(u * 16 + l16) * S_LEN + s] =
            (bf16_t)(acc[u][v] + bcol);
      }
    }
  }
}

// ---------- kernel 2: flash attention, double-buffered async pipeline ----------
// Block = 8 waves = 8 query tiles in one batch. 32-key stages of K (16KB) and
// Vt (16KB) are staged in LDS with async copies; stage i+1 is issued before
// computing stage i, and a partial s_wait_asynccnt(8) releases stage i while
// stage i+1 stays in flight behind the 32 WMMAs of compute.

__global__ void __launch_bounds__(256, 1) flash_attn(
    const bf16_t* __restrict__ Q,   // [B*S, H] bf16, pre-scaled
    const bf16_t* __restrict__ K,   // [B*S, H] bf16
    const bf16_t* __restrict__ Vt,  // [B][H][S] bf16
    float* __restrict__ out) {      // [B*S, H] f32
  __shared__ bf16_t Kl[2 * 32 * 256];          // 2 x 16KB: key-major [32][256]
  __shared__ bf16_t Vl[2 * 256 * 32];          // 2 x 16KB: d-major  [256][32]
  __shared__ bf16_t Pl[8 * 16 * 32];           // 8KB: per-wave P transpose

  const int lane = threadIdx.x & 31;
  const int wave = threadIdx.x >> 5;
  const int l16  = lane & 15;
  const int half = lane >> 4;
  const int c0   = half * 8;
  bf16_t* myp = Pl + wave * (16 * 32);

  int b  = blockIdx.x >> 5;                    // 32 blocks per batch
  int qt = blockIdx.x * 8 + wave;              // 0..1023
  int q0 = (qt & 255) * 16;
  const bf16_t* Qb = Q  + (size_t)b * S_LEN * HDIM;
  const bf16_t* Kb = K  + (size_t)b * S_LEN * HDIM;
  const bf16_t* Vb = Vt + (size_t)b * HDIM * S_LEN;

  // Q tile held entirely in registers as A-fragments (8 x v16bf = 64 VGPRs)
  v16bf aq[8];
  const bf16_t* qr = Qb + (size_t)(q0 + l16) * HDIM;
#pragma unroll
  for (int dc = 0; dc < 8; ++dc)
    aq[dc] = cat16(load8bf(qr + dc * 32 + c0), load8bf(qr + dc * 32 + 16 + c0));

  v8f zero = {};
  float mrow[8], lrow[8];
#pragma unroll
  for (int v = 0; v < 8; ++v) { mrow[v] = -__builtin_inff(); lrow[v] = 0.f; }
  v8f O[16];
#pragma unroll
  for (int t = 0; t < 16; ++t) O[t] = zero;

  const uint32_t kbase = lds_addr(&Kl[0]);
  const uint32_t vbase = lds_addr(&Vl[0]);

  // Issue one stage (8 async b128 per wave): K 16KB contiguous + Vt 256x64B rows.
  auto stage = [&](int buf, int kt) {
    const char* ksrc = (const char*)(Kb + (size_t)kt * HDIM);
    uint32_t kdst = kbase + buf * 16384;
#pragma unroll
    for (int j = 0; j < 4; ++j) {
      int e = threadIdx.x + j * 256;           // 0..1023 16B chunks
      async_copy_b128(kdst + e * 16, ksrc + e * 16);
    }
    const char* vsrc = (const char*)(Vb + (size_t)threadIdx.x * S_LEN + kt);
    uint32_t vdst = vbase + buf * 16384 + threadIdx.x * 64;
#pragma unroll
    for (int j = 0; j < 4; ++j)
      async_copy_b128(vdst + j * 16, vsrc + j * 16);
  };

  stage(0, 0);                                 // prologue: stage 0 in flight

  int i = 0;
  for (int kt = 0; kt < S_LEN; kt += 32, ++i) {
    const int cur = i & 1;
    __syncthreads();           // all waves done with buffer (1-cur) from stage i-1
    if (kt + 32 < S_LEN) {
      stage(1 - cur, kt + 32); // overwrite buffer (1-cur) with stage i+1
      wait_async<8>();         // in-order completion: stage i landed
    } else {
      wait_async<0>();
    }
    __syncthreads();           // stage i visible to every wave

    const bf16_t* Kcur = Kl + cur * (32 * 256);
    const bf16_t* Vcur = Vl + cur * (256 * 32);

    // ---- scores: two 16x16 tiles (keys kt..+15, kt+16..+31), B from LDS ----
    v8f s0 = zero, s1 = zero;
    const bf16_t* kr0 = Kcur + (size_t)l16 * HDIM;
    const bf16_t* kr1 = kr0 + 16 * HDIM;
#pragma unroll
    for (int dc = 0; dc < 8; ++dc) {
      v16bf B0 = cat16(load8bf(kr0 + dc * 32 + c0), load8bf(kr0 + dc * 32 + 16 + c0));
      v16bf B1 = cat16(load8bf(kr1 + dc * 32 + c0), load8bf(kr1 + dc * 32 + 16 + c0));
      s0 = wmma_bf16(aq[dc], B0, s0);
      s1 = wmma_bf16(aq[dc], B1, s1);
    }

    // ---- online softmax (row m = v + half*8; reduce over 16 lanes) ----
    float scv[8];
#pragma unroll
    for (int v = 0; v < 8; ++v) {
      float rmax = fmaxf(s0[v], s1[v]);
#pragma unroll
      for (int off = 1; off < 16; off <<= 1)
        rmax = fmaxf(rmax, __shfl_xor(rmax, off, 32));
      float mnew = fmaxf(mrow[v], rmax);
      float p0 = __expf(s0[v] - mnew);
      float p1 = __expf(s1[v] - mnew);
      float rs = p0 + p1;
#pragma unroll
      for (int off = 1; off < 16; off <<= 1)
        rs += __shfl_xor(rs, off, 32);
      float sc = __expf(mrow[v] - mnew);
      lrow[v] = lrow[v] * sc + rs;
      mrow[v] = mnew;
      scv[v]  = sc;
      int m = v + half * 8;            // C-layout -> row-major LDS (P transpose)
      myp[m * 32 + l16]      = (bf16_t)p0;
      myp[m * 32 + 16 + l16] = (bf16_t)p1;
    }

    // rescale output accumulators
#pragma unroll
    for (int t = 0; t < 16; ++t)
#pragma unroll
      for (int v = 0; v < 8; ++v) O[t][v] *= scv[v];

    // P back as A-fragment (16x32). Same-wave DS ops are in-order: no barrier.
    v16bf Ap = cat16(load8bf(myp + l16 * 32 + c0),
                     load8bf(myp + l16 * 32 + 16 + c0));

    // ---- O += P * V : 16 output d-tiles, B from LDS-staged Vt ----
#pragma unroll
    for (int t = 0; t < 16; ++t) {
      const bf16_t* vr = Vcur + (size_t)(t * 16 + l16) * 32;
      v16bf Bv = cat16(load8bf(vr + c0), load8bf(vr + 16 + c0));
      O[t] = wmma_bf16(Ap, Bv, O[t]);
    }
  }

  // ---- epilogue: normalize by l and store f32 ----
  float* ob = out + (size_t)b * S_LEN * HDIM;
#pragma unroll
  for (int v = 0; v < 8; ++v) {
    float inv = 1.0f / lrow[v];
    int m = v + half * 8;
#pragma unroll
    for (int t = 0; t < 16; ++t)
      ob[(size_t)(q0 + m) * HDIM + t * 16 + l16] = O[t][v] * inv;
  }
}

// ---------- launcher ----------

extern "C" void kernel_launch(void* const* d_in, const int* in_sizes, int n_in,
                              void* d_out, int out_size, void* d_ws, size_t ws_size,
                              hipStream_t stream) {
  const float* x  = (const float*)d_in[0];
  const float* Wq = (const float*)d_in[1];
  const float* bq = (const float*)d_in[2];
  const float* Wk = (const float*)d_in[3];
  const float* bk = (const float*)d_in[4];
  const float* Wv = (const float*)d_in[5];
  const float* bv = (const float*)d_in[6];

  char* ws = (char*)d_ws;
  // layout: Wt (3*256*256 bf16 = 384KB) | Q (8MB) | K (8MB) | Vt (8MB)
  bf16_t* Wt = (bf16_t*)ws;
  bf16_t* Qb = (bf16_t*)(ws + 393216);
  bf16_t* Kb = (bf16_t*)(ws + 393216 + 8388608);
  bf16_t* Vt = (bf16_t*)(ws + 393216 + 2 * 8388608);

  pack_weights<<<768, 256, 0, stream>>>(Wq, Wk, Wv, Wt);
  qkv_proj<<<384, 256, 0, stream>>>(x, Wt, bq, bk, bv, Qb, Kb, Vt);
  flash_attn<<<128, 256, 0, stream>>>(Qb, Kb, Vt, (float*)d_out);
}
